// TopologicalEngine_25907242729864
// MI455X (gfx1250) — matrix operand order repaired and checked by reading, not compiled, not run on hardware
//
#include <hip/hip_runtime.h>
#include <hip/hip_fp16.h>

typedef __attribute__((ext_vector_type(16))) _Float16 v16h;
typedef __attribute__((ext_vector_type(8)))  float    v8f;

// ---------------- problem dims ----------------
constexpr int NCELLS = 65536;
constexpr int DIN = 256, DHID = 512, DOUT = 256, DMID = 128;
constexpr int DCAT = DIN + DHID;   // 768
constexpr int G3   = 3 * DHID;     // 1536
constexpr int ROWS_WG = 64;        // rows per workgroup (4 waves x 16)
constexpr int NFACT = 8;
constexpr int FSIZE = NCELLS / NFACT; // 8192

// ---------------- workspace layout ----------------
// f16 weight region (units: halfs), at byte 0 of d_ws
constexpr size_t H_WA1   = 0;
constexpr size_t H_WG1   = H_WA1  + (size_t)DCAT * DMID;
constexpr size_t H_WA2   = H_WG1  + (size_t)DCAT * DMID;
constexpr size_t H_NWG2  = H_WA2  + (size_t)DMID * DOUT;   // negated Wg2
constexpr size_t H_WIH   = H_NWG2 + (size_t)DMID * DOUT;   // rows 0..255 of W_ih
constexpr size_t H_WHH   = H_WIH  + (size_t)DOUT * G3;
constexpr size_t H_TOTAL = H_WHH  + (size_t)DHID * G3;     // 1,441,792 halfs
// float region (units: floats), fbase = (float*)d_ws + H_TOTAL/2
constexpr size_t F_B2D   = 0;                  // 256: ba2 - bg2
constexpr size_t F_TENS  = F_B2D + DOUT;       // 65536 tensions
constexpr size_t F_FSUM  = F_TENS + NCELLS;    // 8*512 faction sums
constexpr size_t F_COMB  = F_FSUM + (size_t)NFACT * DHID;  // 256 raw weighted out
constexpr size_t F_TMAX  = F_COMB + DOUT;      // tension max (float bits as int)
constexpr size_t F_SUME  = F_TMAX + 1;         // sum exp
constexpr size_t F_TSUM  = F_SUME + 1;         // sum tension
constexpr size_t F_OUT16 = F_TMAX + 4;         // f16 out matrix: NCELLS*DOUT halfs
constexpr size_t F_ZERO_N = (size_t)NFACT * DHID + DOUT + 4; // fsum+comb+scalars

// ---------------- LDS layout (dynamic; starts at LDS offset 0) ----------------
constexpr unsigned L_BSTAGE_BYTES = 2u * 24576u;          // 2 x 24KB B-tile buffers
constexpr unsigned L_BSTAGE_HALFS = L_BSTAGE_BYTES / 2;   // 24576 halfs
constexpr size_t SMEM_BYTES =
    (size_t)L_BSTAGE_BYTES +           // B staging     (48 KB)
    (size_t)ROWS_WG * DCAT * 2 +       // combined f16  (96 KB)
    (size_t)ROWS_WG * DMID * 2 * 2 +   // midA/midG     (32 KB)
    (size_t)ROWS_WG * DOUT * 2 +       // out f16       (32 KB)
    (size_t)ROWS_WG * 4;               // tension       (256 B)  => 213248

// ---------------- WMMA fragment helpers ----------------
__device__ __forceinline__ unsigned lane_id() { return threadIdx.x & 31u; }

__device__ __forceinline__ v8f wmma_f16(v16h a, v16h b, v8f c) {
  return __builtin_amdgcn_wmma_f32_16x16x32_f16(false, a, false, b, (short)0, c,
                                                false, false);
}

// A fragment: 16x32 f16 tile, rows m0..m0+15, K cols k0..k0+31 of row-major [*, ld].
// ISA layout: lanes 0-15 row M=lane, K = {0..7, 16..23}; lanes 16-31 shift K by 8.
__device__ __forceinline__ v16h frag_A(const _Float16* src, int ld, int m0, int k0) {
  const unsigned l = lane_id();
  const unsigned hi = l >> 4;
  const _Float16* p = src + (size_t)(m0 + (int)(l & 15u)) * ld + k0 + (int)hi * 8;
  v16h a;
#pragma unroll
  for (int j = 0; j < 8; ++j) a[j] = p[j];
#pragma unroll
  for (int j = 0; j < 8; ++j) a[j + 8] = p[16 + j];
  return a;
}

// B fragment from an LDS-staged [K x 16] tile (row stride 16 halfs = 32 B):
// lane = N column (lane&15), K = {0..7, 16..23}; lanes 16-31 shift K by 8.
__device__ __forceinline__ v16h frag_B16(const _Float16* base, int k0) {
  const unsigned l = lane_id();
  const _Float16* p = base + (size_t)(k0 + (int)(l >> 4) * 8) * 16 + (int)(l & 15u);
  v16h b;
#pragma unroll
  for (int j = 0; j < 8; ++j) b[j] = p[j * 16];
#pragma unroll
  for (int j = 0; j < 8; ++j) b[j + 8] = p[(16 + j) * 16];
  return b;
}

// C/D layout: element (M = vgpr + 8*(lane>=16), N = lane&15)
__device__ __forceinline__ v8f acc_bias(const float* bias, int n0) {
  const float b = bias[n0 + (int)(lane_id() & 15u)];
  v8f c;
#pragma unroll
  for (int i = 0; i < 8; ++i) c[i] = b;
  return c;
}

__device__ __forceinline__ void acc_store_f16(_Float16* dst, int ld, int m0, int n0,
                                              v8f c) {
  const unsigned l = lane_id();
  const unsigned hi = l >> 4, nn = l & 15u;
#pragma unroll
  for (int vg = 0; vg < 8; ++vg)
    dst[(size_t)(m0 + vg + 8 * (int)hi) * ld + n0 + (int)nn] = (_Float16)c[vg];
}

__device__ __forceinline__ v8f relu8(v8f x) {
  v8f r;
#pragma unroll
  for (int i = 0; i < 8; ++i) r[i] = fmaxf(x[i], 0.0f);
  return r;
}

__device__ __forceinline__ v8f sig8(v8f x) {
  v8f r;
#pragma unroll
  for (int i = 0; i < 8; ++i) r[i] = 1.0f / (1.0f + __expf(-x[i]));
  return r;
}

// ---------------- async B-tile staging (CDNA5 async-LDS path) ----------------
// Copies a [rows x 16]-half tile (col0..col0+15 of row-major W, leading dim ldW)
// into LDS at byte offset ldsByteBase, packed with row stride 32 B. Work is split
// across the 4 waves; each global_load_async_to_lds_b128 moves 32 lanes x 16 B.
// Tracked by ASYNCcnt (rows/64 instructions per wave per call).
__device__ __forceinline__ void stage_tile_async(const _Float16* W, int ldW, int col0,
                                                 int rows, unsigned ldsByteBase) {
  const unsigned wave = threadIdx.x >> 5;
  const unsigned lane = lane_id();
  const int nInst = rows >> 6;  // rows/64 instructions per wave
#pragma unroll
  for (int j = 0; j < nInst; ++j) {
    const unsigned c = (wave * (unsigned)nInst + (unsigned)j) * 32u + lane; // 16B chunk
    const _Float16* src = W + (size_t)(c >> 1) * ldW + col0 + (int)(c & 1u) * 8;
    const unsigned dst = ldsByteBase + c * 16u;
    asm volatile("global_load_async_to_lds_b128 %0, %1, off"
                 :: "v"(dst), "v"(src)
                 : "memory");
  }
}

__device__ __forceinline__ void wait_async_le12() {
  asm volatile("s_wait_asynccnt 0xc" ::: "memory");
}
__device__ __forceinline__ void wait_async_0() {
  asm volatile("s_wait_asynccnt 0x0" ::: "memory");
}

// ---------------- kernels ----------------
__global__ __launch_bounds__(256) void prep_kernel(
    const float* __restrict__ Wa1, const float* __restrict__ Wg1,
    const float* __restrict__ Wa2, const float* __restrict__ Wg2,
    const float* __restrict__ ba2, const float* __restrict__ bg2,
    const float* __restrict__ W_ih, const float* __restrict__ W_hh,
    _Float16* __restrict__ hbase, float* __restrict__ fbase) {
  size_t i = (size_t)blockIdx.x * 256 + threadIdx.x;
  size_t n;
  n = (size_t)DCAT * DMID;
  if (i < n) { hbase[H_WA1 + i]  = (_Float16)Wa1[i];   return; } i -= n;
  if (i < n) { hbase[H_WG1 + i]  = (_Float16)Wg1[i];   return; } i -= n;
  n = (size_t)DMID * DOUT;
  if (i < n) { hbase[H_WA2 + i]  = (_Float16)Wa2[i];   return; } i -= n;
  if (i < n) { hbase[H_NWG2 + i] = (_Float16)(-Wg2[i]); return; } i -= n;
  n = (size_t)DOUT * G3;
  if (i < n) { hbase[H_WIH + i]  = (_Float16)W_ih[i];  return; } i -= n;
  n = (size_t)DHID * G3;
  if (i < n) { hbase[H_WHH + i]  = (_Float16)W_hh[i];  return; } i -= n;
  if (i < (size_t)DOUT) { fbase[F_B2D + i] = ba2[i] - bg2[i]; return; } i -= DOUT;
  if (i < F_ZERO_N) fbase[F_FSUM + i] = 0.0f;   // fsum, comb_raw, tmax/sume/tsum
}

__global__ __launch_bounds__(128) void cell_forward_kernel(
    const float* __restrict__ x, const float* __restrict__ hiddens,
    const float* __restrict__ ba1, const float* __restrict__ bg1,
    const float* __restrict__ W_ih, const float* __restrict__ b_ih,
    const float* __restrict__ b_hh, const _Float16* __restrict__ hbase,
    float* __restrict__ fbase, float* __restrict__ h_out) {
  extern __shared__ char smem[];
  _Float16* bstage = (_Float16*)smem;                   // 2 x 24KB B staging
  _Float16* comb = bstage + L_BSTAGE_HALFS;             // 64 x 768
  _Float16* midA = comb + ROWS_WG * DCAT;               // 64 x 128
  _Float16* midG = midA + ROWS_WG * DMID;               // 64 x 128
  _Float16* outT = midG + ROWS_WG * DMID;               // 64 x 256
  float*    tens = (float*)(outT + ROWS_WG * DOUT);     // 64

  const _Float16* Wa1h  = hbase + H_WA1;
  const _Float16* Wg1h  = hbase + H_WG1;
  const _Float16* Wa2h  = hbase + H_WA2;
  const _Float16* nWg2h = hbase + H_NWG2;
  const _Float16* Wihh  = hbase + H_WIH;
  const _Float16* Whhh  = hbase + H_WHH;
  const float* b2d   = fbase + F_B2D;
  float* tension_g   = fbase + F_TENS;
  float* fsum        = fbase + F_FSUM;
  int*   tmaxb       = (int*)(fbase + F_TMAX);
  _Float16* out16    = (_Float16*)(fbase + F_OUT16);

  const int tid  = (int)threadIdx.x;
  const int row0 = (int)blockIdx.x * ROWS_WG;
  const int m0   = (tid >> 5) * 16;  // wave's 16-row M tile

  // Stage combined = [x | h] into LDS as f16 (coalesced).
  for (int i = tid; i < ROWS_WG * DCAT; i += 128) {
    int r = i / DCAT, c = i - r * DCAT;
    float v = (c < DIN) ? x[c] : hiddens[(size_t)(row0 + r) * DHID + (c - DIN)];
    comb[i] = (_Float16)v;
  }
  __syncthreads();

  // Stage 1: mid = relu(combined @ W1 + b1); W1 16-col slices async-staged to LDS,
  // shared by all 4 waves.
  for (int nt = 0; nt < DMID / 16; ++nt) {
    stage_tile_async(Wa1h, DMID, nt * 16, DCAT, 0u);       // 24 KB
    stage_tile_async(Wg1h, DMID, nt * 16, DCAT, 24576u);   // 24 KB
    wait_async_0();
    __syncthreads();
    v8f accA = acc_bias(ba1, nt * 16);
    v8f accG = acc_bias(bg1, nt * 16);
    for (int kc = 0; kc < DCAT / 32; ++kc) {
      v16h a = frag_A(comb, DCAT, m0, kc * 32);
      accA = wmma_f16(a, frag_B16(bstage, kc * 32), accA);
      accG = wmma_f16(a, frag_B16(bstage + 12288, kc * 32), accG);
    }
    acc_store_f16(midA, DMID, m0, nt * 16, relu8(accA));
    acc_store_f16(midG, DMID, m0, nt * 16, relu8(accG));
    __syncthreads();  // frees bstage for next nt
  }

  // Stage 2: out = midA@Wa2 + midG@(-Wg2) + (ba2-bg2); 16-col slices via async LDS.
  for (int nt = 0; nt < DOUT / 16; ++nt) {
    stage_tile_async(Wa2h, DOUT, nt * 16, DMID, 0u);       // 4 KB
    stage_tile_async(nWg2h, DOUT, nt * 16, DMID, 4096u);   // 4 KB
    wait_async_0();
    __syncthreads();
    v8f acc = acc_bias(b2d, nt * 16);
    for (int kc = 0; kc < DMID / 32; ++kc)
      acc = wmma_f16(frag_A(midA, DMID, m0, kc * 32),
                     frag_B16(bstage, kc * 32), acc);
    for (int kc = 0; kc < DMID / 32; ++kc)
      acc = wmma_f16(frag_A(midG, DMID, m0, kc * 32),
                     frag_B16(bstage + 2048, kc * 32), acc);
    acc_store_f16(outT, DOUT, m0, nt * 16, acc);
    __syncthreads();
  }

  // Stage 3: tension per row, export out & tension, atomic max for softmax.
  if (tid < ROWS_WG) {
    float s = 0.0f;
    for (int c = 0; c < DOUT; ++c) {
      float v = (float)outT[tid * DOUT + c];
      s += v * v;
    }
    s *= (1.0f / DOUT);
    tens[tid] = s;
    tension_g[row0 + tid] = s;
    atomicMax(tmaxb, __float_as_int(s));  // tensions >= 0: int order == float order
  }
  {  // copy out tile to global (u32 chunks, coalesced)
    const unsigned* s32 = (const unsigned*)outT;
    unsigned* d32 = (unsigned*)out16 + (size_t)row0 * (DOUT / 2);
    for (int i = tid; i < ROWS_WG * DOUT / 2; i += 128) d32[i] = s32[i];
  }
  __syncthreads();

  // Stage 4: GRU (gates [r|z|n] over 1536 cols), 96 slots = 32 col-tiles x 3 gates.
  // Per slot: W_ih slice (256x16, 8KB) + W_hh slice (512x16, 16KB) double-buffered
  // in LDS via async loads; all 4 waves consume the same tiles.
  {
    // prefetch slot 0
    stage_tile_async(Wihh, G3, 0, DOUT, 0u);
    stage_tile_async(Whhh, G3, 0, DHID, 8192u);
    v8f r = {}, z = {};
    for (int s = 0; s < 96; ++s) {
      if (s + 1 < 96) {
        const int nt1 = (s + 1) / 3, g1 = (s + 1) - nt1 * 3;
        const int gcol1 = g1 * DHID + nt1 * 16;
        const unsigned base1 = (unsigned)(((s + 1) & 1) * 24576);
        stage_tile_async(Wihh, G3, gcol1, DOUT, base1);
        stage_tile_async(Whhh, G3, gcol1, DHID, base1 + 8192u);
        wait_async_le12();   // slot s's 12 loads complete (in-order)
      } else {
        wait_async_0();
      }
      __syncthreads();  // slot-s tiles visible to all waves

      const int nt = s / 3, g = s - nt * 3;
      const int cH = nt * 16, gcol = g * DHID + cH;
      const _Float16* bi = bstage + (size_t)(s & 1) * 12288;  // 256x16
      const _Float16* bh = bi + 4096;                         // 512x16

      // i-gate: K=257 (8x32 WMMA over out + rank-1 tension term)
      v8f gi = acc_bias(b_ih, gcol);
      for (int kc = 0; kc < DOUT / 32; ++kc)
        gi = wmma_f16(frag_A(outT, DOUT, m0, kc * 32), frag_B16(bi, kc * 32), gi);
      {
        const unsigned l = lane_id();
        const unsigned hi = l >> 4;
        const float wl = W_ih[(size_t)DOUT * G3 + gcol + (int)(l & 15u)]; // row 256
#pragma unroll
        for (int vg = 0; vg < 8; ++vg)
          gi[vg] += tens[m0 + vg + 8 * (int)hi] * wl;
      }
      // h-gate: K=512 from h-portion of combined LDS tile
      v8f gh = acc_bias(b_hh, gcol);
      for (int kc = 0; kc < DHID / 32; ++kc)
        gh = wmma_f16(frag_A(comb, DCAT, m0, DIN + kc * 32),
                      frag_B16(bh, kc * 32), gh);

      if (g == 0) {
        r = sig8(gi + gh);
      } else if (g == 1) {
        z = sig8(gi + gh);
      } else {
        const unsigned l = lane_id();
        const unsigned hi = l >> 4, nn = l & 15u;
        float fpart = 0.0f;
#pragma unroll
        for (int vg = 0; vg < 8; ++vg) {
          int M = m0 + vg + 8 * (int)hi;
          float ng = tanhf(gi[vg] + r[vg] * gh[vg]);
          float hv = (float)comb[(size_t)M * DCAT + DIN + cH + (int)nn];
          float nh = (1.0f - z[vg]) * ng + z[vg] * hv;
          h_out[(size_t)(row0 + M) * DHID + cH + (int)nn] = nh;
          fpart += nh;
        }
        atomicAdd(&fsum[(row0 >> 13) * DHID + cH + (int)nn], fpart);
      }
      __syncthreads();  // all waves done with buf (s&1); safe to refill at s+2
    }
  }
}

__global__ __launch_bounds__(256) void softmax_combine_kernel(float* __restrict__ fbase) {
  __shared__ float wl[256];
  __shared__ float swsum, stsum;
  const float* tension = fbase + F_TENS;
  const _Float16* out16 = (const _Float16*)(fbase + F_OUT16);
  const int* tmaxb = (const int*)(fbase + F_TMAX);
  float* comb = fbase + F_COMB;

  const int tid = (int)threadIdx.x;
  const int r0 = (int)blockIdx.x * 256;
  if (tid == 0) { swsum = 0.0f; stsum = 0.0f; }
  __syncthreads();
  const float tmax = __int_as_float(*tmaxb);
  const float t = tension[r0 + tid];
  const float w = __expf(t - tmax);
  wl[tid] = w;
  atomicAdd(&swsum, w);
  atomicAdd(&stsum, t);
  __syncthreads();
  float acc = 0.0f;
  for (int r = 0; r < 256; ++r)
    acc += wl[r] * (float)out16[(size_t)(r0 + r) * DOUT + tid];
  atomicAdd(&comb[tid], acc);
  if (tid == 0) {
    atomicAdd(fbase + F_SUME, swsum);
    atomicAdd(fbase + F_TSUM, stsum);
  }
}

__global__ __launch_bounds__(256) void faction_finalize_kernel(
    float* __restrict__ h, const float* __restrict__ fbase, const int* __restrict__ step) {
  const float* fsum = fbase + F_FSUM;
  const size_t idx = (size_t)blockIdx.x * 256 + threadIdx.x;
  const int col = (int)(idx & (DHID - 1));
  const int row = (int)(idx >> 9);
  const int f = row >> 13;  // /8192
  const float fmean = fsum[f * DHID + col] * (1.0f / FSIZE);
  float hs = 0.85f * h[idx] + 0.15f * fmean;
  if (*step > 5 && (row & (FSIZE - 1)) < FSIZE / 4) {
    float g = 0.0f;
    for (int ff = 0; ff < NFACT; ++ff) g += fsum[ff * DHID + col];
    hs = 0.85f * hs + 0.15f * (g * (1.0f / NCELLS));  // global mean of synced h
  }
  h[idx] = hs;
}

__global__ void finalize_out_kernel(const float* __restrict__ fbase,
                                    float* __restrict__ out) {
  const int i = (int)blockIdx.x * 256 + (int)threadIdx.x;
  if (i < DOUT)       out[i] = fbase[F_COMB + i] / fbase[F_SUME];
  else if (i == DOUT) out[DOUT] = fbase[F_TSUM] * (1.0f / NCELLS);
}

// ---------------- launcher ----------------
extern "C" void kernel_launch(void* const* d_in, const int* in_sizes, int n_in,
                              void* d_out, int out_size, void* d_ws, size_t ws_size,
                              hipStream_t stream) {
  const float* x    = (const float*)d_in[0];
  const float* hid  = (const float*)d_in[1];
  const float* Wa1  = (const float*)d_in[2];
  const float* ba1  = (const float*)d_in[3];
  const float* Wa2  = (const float*)d_in[4];
  const float* ba2  = (const float*)d_in[5];
  const float* Wg1  = (const float*)d_in[6];
  const float* bg1  = (const float*)d_in[7];
  const float* Wg2  = (const float*)d_in[8];
  const float* bg2  = (const float*)d_in[9];
  const float* W_ih = (const float*)d_in[10];
  const float* b_ih = (const float*)d_in[11];
  const float* W_hh = (const float*)d_in[12];
  const float* b_hh = (const float*)d_in[13];
  const int*   step = (const int*)d_in[14];

  _Float16* hbase = (_Float16*)d_ws;
  float* fbase = (float*)d_ws + H_TOTAL / 2;
  float* out   = (float*)d_out;
  float* h_out = out + DOUT + 1;  // h_final region

  {
    const size_t total = H_TOTAL + (size_t)DOUT + F_ZERO_N;
    const int blocks = (int)((total + 255) / 256);
    prep_kernel<<<blocks, 256, 0, stream>>>(Wa1, Wg1, Wa2, Wg2, ba2, bg2, W_ih,
                                            W_hh, hbase, fbase);
  }
  (void)hipFuncSetAttribute((const void*)cell_forward_kernel,
                            hipFuncAttributeMaxDynamicSharedMemorySize,
                            (int)SMEM_BYTES);
  cell_forward_kernel<<<NCELLS / ROWS_WG, 128, SMEM_BYTES, stream>>>(
      x, hid, ba1, bg1, W_ih, b_ih, b_hh, hbase, fbase, h_out);
  softmax_combine_kernel<<<NCELLS / 256, 256, 0, stream>>>(fbase);
  faction_finalize_kernel<<<(int)(((size_t)NCELLS * DHID) / 256), 256, 0, stream>>>(
      h_out, fbase, step);
  finalize_out_kernel<<<2, 256, 0, stream>>>(fbase, out);
}
